// Cross_Fourier_Fusion_Conditioner_86500641341938
// MI455X (gfx1250) — matrix operand order repaired and checked by reading, not compiled, not run on hardware
//
#include <hip/hip_runtime.h>
#include <hip/hip_bf16.h>
#include <math.h>
#include <stdint.h>

// ---------------------------------------------------------------------------
// Cross_Fourier_Fusion_Conditioner for MI455X (gfx1250, wave32, WMMA).
// GEMMs: v_wmma_f32_16x16x32_bf16, double-buffered LDS tiles streamed with
// global_load_async_to_lds_b128 (ASYNCcnt) when available.
// ---------------------------------------------------------------------------

typedef __bf16 bf16;
typedef __attribute__((ext_vector_type(16))) __bf16        v16bf;
typedef __attribute__((ext_vector_type(8)))  float         v8f;
typedef __attribute__((ext_vector_type(4)))  unsigned int  u32x4;
typedef __attribute__((ext_vector_type(4)))  int           v4i;

#define B_    32
#define T_    256
#define OBS_  256
#define DIM_  512
#define H_    8
#define HD_   64
#define HID_  2048
#define MROWS (B_ * T_)   // 8192

#if __has_builtin(__builtin_amdgcn_global_load_async_to_lds_b128)
#define HAVE_ASYNC 1
#else
#define HAVE_ASYNC 0
#endif

#if HAVE_ASYNC
static __device__ __forceinline__ void async_cp16(const void* g, void* l) {
  // Builtin expects (v4i AS1*, v4i AS3*, int offset, int cpol); cast through
  // uintptr_t (LDS aperture low 32 bits == LDS offset, so this is exact).
  auto gp = reinterpret_cast<__attribute__((address_space(1))) v4i*>(
      reinterpret_cast<uintptr_t>(g));
  auto lp = reinterpret_cast<__attribute__((address_space(3))) v4i*>(
      reinterpret_cast<uintptr_t>(l));
  __builtin_amdgcn_global_load_async_to_lds_b128(gp, lp, 0, 0);
}
static __device__ __forceinline__ void wait_async0() {
#if __has_builtin(__builtin_amdgcn_s_wait_asynccnt)
  __builtin_amdgcn_s_wait_asynccnt(0);
#else
  asm volatile("s_wait_asynccnt 0x0" ::: "memory");
#endif
}
#endif

// ---------------- f32 -> bf16 with transpose: W[K,N] -> Wt[N,K] ------------
// LDS-tiled 32x32 so both global sides stay coalesced.
__global__ __launch_bounds__(256) void f2bf_t_kernel(const float* __restrict__ in,
                                                     bf16* __restrict__ out,
                                                     int K, int N) {
  __shared__ bf16 tile[32][33];
  const int kb = blockIdx.x * 32;
  const int nb = blockIdx.y * 32;
  const int tx = threadIdx.x & 31;
  const int ty = threadIdx.x >> 5;   // 0..7
#pragma unroll
  for (int j = 0; j < 32; j += 8)
    tile[ty + j][tx] = (bf16)in[(size_t)(kb + ty + j) * N + nb + tx];
  __syncthreads();
#pragma unroll
  for (int j = 0; j < 32; j += 8)
    out[(size_t)(nb + ty + j) * K + kb + tx] = tile[tx][ty + j];
}

// ---------------- DFT over time axis -> magnitude / phase (bf16) -----------
__global__ __launch_bounds__(256) void dft_kernel(const float* __restrict__ x,
                                                  bf16* __restrict__ mag,
                                                  bf16* __restrict__ ph) {
  __shared__ float ct[T_], st[T_];
  int bt = blockIdx.x;
  int b  = bt >> 8, tp = bt & 255;
  int o  = threadIdx.x;
  float ang = -6.283185307179586f * (float)o * (1.0f / 256.0f);
  float s, c;
  sincosf(ang, &s, &c);
  ct[o] = c; st[o] = s;
  __syncthreads();
  const float* xb = x + ((size_t)b * T_) * OBS_ + o;
  float re = 0.f, im = 0.f;
#pragma unroll 4
  for (int t = 0; t < T_; ++t) {
    int w = (tp * t) & 255;
    float v = xb[(size_t)t * OBS_];
    re = fmaf(v, ct[w], re);
    im = fmaf(v, st[w], im);
  }
  size_t oi = (size_t)bt * OBS_ + o;
  mag[oi] = (bf16)sqrtf(re * re + im * im);
  ph[oi]  = (bf16)atan2f(im, re);
}

// ---------------- polar -> rectangular (in place, f32) ---------------------
__global__ void polar_kernel(float* __restrict__ m_io, float* __restrict__ p_io, int n) {
  int i = blockIdx.x * blockDim.x + threadIdx.x;
  if (i < n) {
    float m = m_io[i], p = p_io[i];
    float s, c;
    sincosf(p, &s, &c);
    m_io[i] = m * c;   // real
    p_io[i] = m * s;   // imag
  }
}

// ---------------- inverse DFT (real part) + residual -> bf16 ---------------
__global__ __launch_bounds__(256) void idft_kernel(const float* __restrict__ fre,
                                                   const float* __restrict__ fim,
                                                   const float* __restrict__ xres,
                                                   bf16* __restrict__ out) {
  __shared__ float ct[T_], st[T_];
  int bt = blockIdx.x;
  int b  = bt >> 8, t = bt & 255;
  int o  = threadIdx.x;
  float ang = 6.283185307179586f * (float)o * (1.0f / 256.0f);
  float s, c;
  sincosf(ang, &s, &c);
  ct[o] = c; st[o] = s;
  __syncthreads();
  const float* rb = fre + ((size_t)b * T_) * OBS_ + o;
  const float* ib = fim + ((size_t)b * T_) * OBS_ + o;
  float acc = 0.f;
#pragma unroll 4
  for (int tp = 0; tp < T_; ++tp) {
    int w = (t * tp) & 255;
    acc = fmaf(rb[(size_t)tp * OBS_],  ct[w], acc);
    acc = fmaf(ib[(size_t)tp * OBS_], -st[w], acc);
  }
  size_t oi = (size_t)bt * OBS_ + o;
  out[oi] = (bf16)(acc * (1.0f / 256.0f) + xres[oi]);
}

// ---------------- bf16 WMMA GEMM: C = act(A[M,K] @ W + bias) ---------------
// W supplied PRE-TRANSPOSED as Wt[N,K] so all LDS staging is b128 copies.
// Block: 256 threads = 8 waves (2x4 over a 64x128 tile); wave owns 32x32 =
// 2x2 WMMA accumulators. BK=64 (two WMMA K-sub-steps per staged tile),
// double-buffered LDS streamed via global_load_async_to_lds_b128.
//   A frag (16x32 bf16): lane<16 -> M=lane,    elems 0..7 = K 0..7, 8..15 = K 16..23
//                        lane>=16 -> M=lane-16, K halves 8..15 / 24..31
//   B frag (32x16 bf16): lane<16 -> N=lane,    elems 0..7 = K 0..7, 8..15 = K 16..23
//                        lane>=16 -> N=lane-16, K halves 8..15 / 24..31
#define BM_   64
#define BN_   128
#define BK_   64
#define LDT_  72   // elems; 144B row stride keeps 16B alignment

template <bool RELU, bool OUTBF>
__global__ __launch_bounds__(256) void gemm_kernel(const bf16* __restrict__ A,
                                                   const bf16* __restrict__ Wt,
                                                   const float* __restrict__ bias,
                                                   void* __restrict__ Cout,
                                                   int M, int N, int K) {
  __shared__ __align__(16) bf16 sA[2][BM_ * LDT_];
  __shared__ __align__(16) bf16 sB[2][BN_ * LDT_];

  const int m0   = blockIdx.x * BM_;
  const int n0   = blockIdx.y * BN_;
  const int tid  = threadIdx.x;
  const int wave = tid >> 5;
  const int lane = tid & 31;
  const int wm   = wave & 1;        // 0..1  -> 32-row slice
  const int wn   = wave >> 1;       // 0..3  -> 32-col slice
  const int l16  = lane & 15;
  const int hi   = lane >> 4;       // 0 or 1
  const int akb  = hi * 8;          // A K-base for this half-wave
  const int bkb  = hi * 8;          // B K-base for this half-wave

  v8f acc[2][2] = {};

  // staging maps (all b128-granular)
  const int arow = tid >> 2;            // 0..63
  const int acol = (tid & 3) * 16;      // 0,16,32,48
  const int brow = tid >> 1;            // 0..127 (N)
  const int bcol = (tid & 1) * 32;      // 0,32   (K)

  auto stage = [&](int sbuf, int k0s) {
    const bf16* ga = A  + (size_t)(m0 + arow) * K + (k0s + acol);
    bf16*       la = &sA[sbuf][arow * LDT_ + acol];
    const bf16* gb = Wt + (size_t)(n0 + brow) * K + (k0s + bcol);
    bf16*       lb = &sB[sbuf][brow * LDT_ + bcol];
#if HAVE_ASYNC
    async_cp16(ga,      la);
    async_cp16(ga + 8,  la + 8);
    async_cp16(gb,      lb);
    async_cp16(gb + 8,  lb + 8);
    async_cp16(gb + 16, lb + 16);
    async_cp16(gb + 24, lb + 24);
#else
    *(u32x4*)la        = *(const u32x4*)ga;
    *(u32x4*)(la + 8)  = *(const u32x4*)(ga + 8);
    *(u32x4*)lb        = *(const u32x4*)gb;
    *(u32x4*)(lb + 8)  = *(const u32x4*)(gb + 8);
    *(u32x4*)(lb + 16) = *(const u32x4*)(gb + 16);
    *(u32x4*)(lb + 24) = *(const u32x4*)(gb + 24);
#endif
  };

  stage(0, 0);
  int buf = 0;
  for (int k0 = 0; k0 < K; k0 += BK_) {
#if HAVE_ASYNC
    wait_async0();                 // our async writes to LDS are complete
#endif
    __syncthreads();               // everyone's tile data is visible
    if (k0 + BK_ < K) {
      stage(buf ^ 1, k0 + BK_);    // stream next tile into the other buffer
      if (k0 + 2 * BK_ < K) {      // warm L2 for the tile after next
        __builtin_prefetch(A  + (size_t)(m0 + arow) * K + k0 + 2 * BK_ + acol, 0, 1);
        __builtin_prefetch(Wt + (size_t)(n0 + brow) * K + k0 + 2 * BK_ + bcol, 0, 1);
      }
    }

#pragma unroll
    for (int kk = 0; kk < BK_; kk += 32) {
      union FragU { v16bf v; u32x4 q[2]; };
      FragU fa[2], fb[2];
#pragma unroll
      for (int mt = 0; mt < 2; ++mt) {
        const bf16* base = &sA[buf][(wm * 32 + mt * 16 + l16) * LDT_ + kk + akb];
        fa[mt].q[0] = *(const u32x4*)base;          // K = akb .. akb+7
        fa[mt].q[1] = *(const u32x4*)(base + 16);   // K = akb+16 .. akb+23
      }
#pragma unroll
      for (int nt = 0; nt < 2; ++nt) {
        const bf16* base = &sB[buf][(wn * 32 + nt * 16 + l16) * LDT_ + kk + bkb];
        fb[nt].q[0] = *(const u32x4*)base;          // K = bkb .. bkb+7
        fb[nt].q[1] = *(const u32x4*)(base + 16);   // K = bkb+16 .. bkb+23
      }
#pragma unroll
      for (int mt = 0; mt < 2; ++mt)
#pragma unroll
        for (int nt = 0; nt < 2; ++nt)
          acc[mt][nt] = __builtin_amdgcn_wmma_f32_16x16x32_bf16(
              false, fa[mt].v, false, fb[nt].v, (short)0, acc[mt][nt], false, false);
    }
    buf ^= 1;
  }

  // ---- epilogue: C/D layout: VGPR r -> row = r + hi*8, col = l16
#pragma unroll
  for (int mt = 0; mt < 2; ++mt) {
#pragma unroll
    for (int nt = 0; nt < 2; ++nt) {
      const int ncol = n0 + wn * 32 + nt * 16 + l16;
      const float bv = bias[ncol];
#pragma unroll
      for (int r = 0; r < 8; ++r) {
        const int mrow = m0 + wm * 32 + mt * 16 + hi * 8 + r;
        float v = acc[mt][nt][r] + bv;
        if (RELU) v = fmaxf(v, 0.0f);
        if (OUTBF) ((bf16*)Cout)[(size_t)mrow * N + ncol] = (bf16)v;
        else       ((float*)Cout)[(size_t)mrow * N + ncol] = v;
      }
    }
  }
}

// ---------------- head-axis attention per (b, t) ---------------------------
// Faithful split_heads scramble: x[b,t,h,d] = mat[b*T + (t&3)*64+d, h*64 + (t>>2)]
__global__ __launch_bounds__(256) void attn_kernel(const bf16* __restrict__ Q,
                                                   const bf16* __restrict__ Km,
                                                   const bf16* __restrict__ VH,
                                                   const bf16* __restrict__ VL,
                                                   float* __restrict__ out) {
  __shared__ float qs[H_][HD_], ks[H_][HD_], vhs[H_][HD_], vls[H_][HD_];
  __shared__ float pr[H_][H_];
  __shared__ float sc[H_][H_];

  const int bt  = blockIdx.x;
  const int b   = bt >> 8, t = bt & 255;
  const int tid = threadIdx.x;

  for (int i = tid; i < H_ * HD_; i += 256) {
    int h = i >> 6, d = i & 63;
    size_t row = (size_t)b * T_ + ((t & 3) * 64 + d);
    size_t idx = row * DIM_ + h * 64 + (t >> 2);
    qs[h][d]  = (float)Q[idx];
    ks[h][d]  = (float)Km[idx];
    vhs[h][d] = (float)VH[idx];
    vls[h][d] = (float)VL[idx];
  }
  __syncthreads();

  if (tid < H_ * H_) {
    int h = tid >> 3, g = tid & 7;
    float a = 0.f;
#pragma unroll
    for (int d = 0; d < HD_; ++d) a = fmaf(qs[h][d], ks[g][d], a);
    sc[h][g] = a * 0.125f;  // 1/sqrt(HD)
  }
  __syncthreads();

  if (tid < H_) {
    float mx = sc[tid][0];
#pragma unroll
    for (int g = 1; g < H_; ++g) mx = fmaxf(mx, sc[tid][g]);
    float e[H_], sum = 0.f;
#pragma unroll
    for (int g = 0; g < H_; ++g) { e[g] = expf(sc[tid][g] - mx); sum += e[g]; }
    float inv = 1.0f / sum;
#pragma unroll
    for (int g = 0; g < H_; ++g) pr[tid][g] = e[g] * inv;
  }
  __syncthreads();

  const size_t OOFF = (size_t)B_ * H_ * T_ * HD_;
  for (int i = tid; i < H_ * HD_; i += 256) {
    int h = i >> 6, d = i & 63;
    float oh = 0.f, ol = 0.f;
#pragma unroll
    for (int g = 0; g < H_; ++g) {
      float p = pr[h][g];
      oh = fmaf(p, vhs[g][d], oh);
      ol = fmaf(p, vls[g][d], ol);
    }
    size_t oidx = (((size_t)b * H_ + h) * T_ + t) * HD_ + d;
    out[oidx]        = oh;
    out[OOFF + oidx] = ol;
  }
}

// ---------------------------------------------------------------------------
// Host orchestration.  Input order = JAX key-sorted pytree flatten:
//  0 high_freq, 1 low_freq, then params dict sorted:
//  fh_mag, fh_ph, fl_mag, fl_ph, high, k, low, q, v; each layer {b, w}.
// ---------------------------------------------------------------------------
enum {
  IN_HF = 0, IN_LF = 1,
  FHM_B0 = 2,  FHM_W0 = 3,  FHM_B1 = 4,  FHM_W1 = 5,
  FHP_B0 = 6,  FHP_W0 = 7,  FHP_B1 = 8,  FHP_W1 = 9,
  FLM_B0 = 10, FLM_W0 = 11, FLM_B1 = 12, FLM_W1 = 13,
  FLP_B0 = 14, FLP_W0 = 15, FLP_B1 = 16, FLP_W1 = 17,
  HI_B0 = 18, HI_W0 = 19, HI_B1 = 20, HI_W1 = 21, HI_B2 = 22, HI_W2 = 23,
  K_B0 = 24, K_W0 = 25, K_B1 = 26, K_W1 = 27,
  LO_B0 = 28, LO_W0 = 29, LO_B1 = 30, LO_W1 = 31, LO_B2 = 32, LO_W2 = 33,
  Q_B0 = 34, Q_W0 = 35, Q_B1 = 36, Q_W1 = 37,
  V_B0 = 38, V_W0 = 39, V_B1 = 40, V_W1 = 41,
};

extern "C" void kernel_launch(void* const* d_in, const int* in_sizes, int n_in,
                              void* d_out, int out_size, void* d_ws, size_t ws_size,
                              hipStream_t stream) {
  (void)in_sizes; (void)n_in; (void)out_size; (void)ws_size;

  char* ws = (char*)d_ws;
  size_t off = 0;
  auto alloc = [&](size_t bytes) -> void* {
    void* p = ws + off;
    off += (bytes + 255) & ~(size_t)255;
    return p;
  };

  // bf16, pre-transposed weight copies: Wt[N,K]
  bf16* w_fhm0 = (bf16*)alloc((size_t)OBS_ * HID_ * 2);
  bf16* w_fhm1 = (bf16*)alloc((size_t)HID_ * OBS_ * 2);
  bf16* w_fhp0 = (bf16*)alloc((size_t)OBS_ * HID_ * 2);
  bf16* w_fhp1 = (bf16*)alloc((size_t)HID_ * OBS_ * 2);
  bf16* w_flm0 = (bf16*)alloc((size_t)OBS_ * HID_ * 2);
  bf16* w_flm1 = (bf16*)alloc((size_t)HID_ * OBS_ * 2);
  bf16* w_flp0 = (bf16*)alloc((size_t)OBS_ * HID_ * 2);
  bf16* w_flp1 = (bf16*)alloc((size_t)HID_ * OBS_ * 2);
  bf16* w_hi0  = (bf16*)alloc((size_t)OBS_ * DIM_ * 2);
  bf16* w_hi1  = (bf16*)alloc((size_t)DIM_ * HID_ * 2);
  bf16* w_hi2  = (bf16*)alloc((size_t)HID_ * DIM_ * 2);
  bf16* w_lo0  = (bf16*)alloc((size_t)OBS_ * DIM_ * 2);
  bf16* w_lo1  = (bf16*)alloc((size_t)DIM_ * HID_ * 2);
  bf16* w_lo2  = (bf16*)alloc((size_t)HID_ * DIM_ * 2);
  bf16* w_q0   = (bf16*)alloc((size_t)DIM_ * HID_ * 2);
  bf16* w_q1   = (bf16*)alloc((size_t)HID_ * DIM_ * 2);
  bf16* w_k0   = (bf16*)alloc((size_t)DIM_ * HID_ * 2);
  bf16* w_k1   = (bf16*)alloc((size_t)HID_ * DIM_ * 2);
  bf16* w_v0   = (bf16*)alloc((size_t)DIM_ * HID_ * 2);
  bf16* w_v1   = (bf16*)alloc((size_t)HID_ * DIM_ * 2);

  // activations
  bf16*  MAG  = (bf16*)alloc((size_t)MROWS * OBS_ * 2);
  bf16*  PH   = (bf16*)alloc((size_t)MROWS * OBS_ * 2);
  bf16*  HBUF = (bf16*)alloc((size_t)MROWS * HID_ * 2);
  float* FRE  = (float*)alloc((size_t)MROWS * OBS_ * 4);
  float* FIM  = (float*)alloc((size_t)MROWS * OBS_ * 4);
  bf16*  HF   = (bf16*)alloc((size_t)MROWS * OBS_ * 2);
  bf16*  LF   = (bf16*)alloc((size_t)MROWS * OBS_ * 2);
  bf16*  X512 = (bf16*)alloc((size_t)MROWS * DIM_ * 2);
  bf16*  HF3  = (bf16*)alloc((size_t)MROWS * DIM_ * 2);
  bf16*  LF3  = (bf16*)alloc((size_t)MROWS * DIM_ * 2);
  bf16*  QB   = (bf16*)alloc((size_t)MROWS * DIM_ * 2);
  bf16*  KB   = (bf16*)alloc((size_t)MROWS * DIM_ * 2);
  bf16*  VHB  = (bf16*)alloc((size_t)MROWS * DIM_ * 2);
  bf16*  VLB  = (bf16*)alloc((size_t)MROWS * DIM_ * 2);

  // convert + transpose a weight: in [K,N] f32 -> out [N,K] bf16
  auto convT = [&](int idx, bf16* dst, int K, int N) {
    dim3 g(K / 32, N / 32);
    f2bf_t_kernel<<<g, 256, 0, stream>>>((const float*)d_in[idx], dst, K, N);
  };
  auto gemm = [&](const bf16* A, const bf16* Wt, int bidx, void* C,
                  int M, int N, int K, bool relu, bool outbf) {
    const float* bias = (const float*)d_in[bidx];
    dim3 g(M / BM_, N / BN_);
    if (relu && outbf)
      gemm_kernel<true, true><<<g, 256, 0, stream>>>(A, Wt, bias, C, M, N, K);
    else if (!relu && outbf)
      gemm_kernel<false, true><<<g, 256, 0, stream>>>(A, Wt, bias, C, M, N, K);
    else if (relu && !outbf)
      gemm_kernel<true, false><<<g, 256, 0, stream>>>(A, Wt, bias, C, M, N, K);
    else
      gemm_kernel<false, false><<<g, 256, 0, stream>>>(A, Wt, bias, C, M, N, K);
  };

  // -- weight conversion + transpose (every call; deterministic)
  convT(FHM_W0, w_fhm0, OBS_, HID_);  convT(FHM_W1, w_fhm1, HID_, OBS_);
  convT(FHP_W0, w_fhp0, OBS_, HID_);  convT(FHP_W1, w_fhp1, HID_, OBS_);
  convT(FLM_W0, w_flm0, OBS_, HID_);  convT(FLM_W1, w_flm1, HID_, OBS_);
  convT(FLP_W0, w_flp0, OBS_, HID_);  convT(FLP_W1, w_flp1, HID_, OBS_);
  convT(HI_W0, w_hi0, OBS_, DIM_);    convT(HI_W1, w_hi1, DIM_, HID_);
  convT(HI_W2, w_hi2, HID_, DIM_);
  convT(LO_W0, w_lo0, OBS_, DIM_);    convT(LO_W1, w_lo1, DIM_, HID_);
  convT(LO_W2, w_lo2, HID_, DIM_);
  convT(Q_W0, w_q0, DIM_, HID_);      convT(Q_W1, w_q1, HID_, DIM_);
  convT(K_W0, w_k0, DIM_, HID_);      convT(K_W1, w_k1, HID_, DIM_);
  convT(V_W0, w_v0, DIM_, HID_);      convT(V_W1, w_v1, HID_, DIM_);

  const int NE = MROWS * OBS_;

  // -- high stream: FFT -> MLP(mag), MLP(ph) -> iFFT + residual
  dft_kernel<<<MROWS, 256, 0, stream>>>((const float*)d_in[IN_HF], MAG, PH);
  gemm(MAG, w_fhm0, FHM_B0, HBUF, MROWS, HID_, OBS_, true, true);
  gemm(HBUF, w_fhm1, FHM_B1, FRE, MROWS, OBS_, HID_, false, false);
  gemm(PH, w_fhp0, FHP_B0, HBUF, MROWS, HID_, OBS_, true, true);
  gemm(HBUF, w_fhp1, FHP_B1, FIM, MROWS, OBS_, HID_, false, false);
  polar_kernel<<<(NE + 255) / 256, 256, 0, stream>>>(FRE, FIM, NE);
  idft_kernel<<<MROWS, 256, 0, stream>>>(FRE, FIM, (const float*)d_in[IN_HF], HF);

  // -- low stream
  dft_kernel<<<MROWS, 256, 0, stream>>>((const float*)d_in[IN_LF], MAG, PH);
  gemm(MAG, w_flm0, FLM_B0, HBUF, MROWS, HID_, OBS_, true, true);
  gemm(HBUF, w_flm1, FLM_B1, FRE, MROWS, OBS_, HID_, false, false);
  gemm(PH, w_flp0, FLP_B0, HBUF, MROWS, HID_, OBS_, true, true);
  gemm(HBUF, w_flp1, FLP_B1, FIM, MROWS, OBS_, HID_, false, false);
  polar_kernel<<<(NE + 255) / 256, 256, 0, stream>>>(FRE, FIM, NE);
  idft_kernel<<<MROWS, 256, 0, stream>>>(FRE, FIM, (const float*)d_in[IN_LF], LF);

  // -- high / low projection MLPs: 256 -> 512 -> 2048 -> 512
  gemm(HF, w_hi0, HI_B0, X512, MROWS, DIM_, OBS_, true, true);
  gemm(X512, w_hi1, HI_B1, HBUF, MROWS, HID_, DIM_, true, true);
  gemm(HBUF, w_hi2, HI_B2, HF3, MROWS, DIM_, HID_, false, true);
  gemm(LF, w_lo0, LO_B0, X512, MROWS, DIM_, OBS_, true, true);
  gemm(X512, w_lo1, LO_B1, HBUF, MROWS, HID_, DIM_, true, true);
  gemm(HBUF, w_lo2, LO_B2, LF3, MROWS, DIM_, HID_, false, true);

  // -- q / k / v MLPs: 512 -> 2048 -> 512
  gemm(HF3, w_q0, Q_B0, HBUF, MROWS, HID_, DIM_, true, true);
  gemm(HBUF, w_q1, Q_B1, QB, MROWS, DIM_, HID_, false, true);
  gemm(LF3, w_k0, K_B0, HBUF, MROWS, HID_, DIM_, true, true);
  gemm(HBUF, w_k1, K_B1, KB, MROWS, DIM_, HID_, false, true);
  gemm(HF3, w_v0, V_B0, HBUF, MROWS, HID_, DIM_, true, true);
  gemm(HBUF, w_v1, V_B1, VHB, MROWS, DIM_, HID_, false, true);
  gemm(LF3, w_v0, V_B0, HBUF, MROWS, HID_, DIM_, true, true);
  gemm(HBUF, w_v1, V_B1, VLB, MROWS, DIM_, HID_, false, true);

  // -- head-axis attention, writes both outputs
  attn_kernel<<<MROWS, 256, 0, stream>>>(QB, KB, VHB, VLB, (float*)d_out);
}
